// CausalAttention_3891240370859
// MI455X (gfx1250) — matrix operand order repaired and checked by reading, not compiled, not run on hardware
//
#include <hip/hip_runtime.h>

typedef __attribute__((ext_vector_type(16))) _Float16 v16h;
typedef __attribute__((ext_vector_type(8)))  _Float16 v8h;
typedef __attribute__((ext_vector_type(4)))  _Float16 v4h;
typedef __attribute__((ext_vector_type(8)))  float    v8f;
typedef __attribute__((ext_vector_type(4)))  float    v4f;

#define B_ 4
#define S_ 2048
#define D_ 1024
#define NEGBIG (-3.0e38f)

// Padded LDS strides (in halves) to break 64-bank conflicts.
// 40 halves = 80 B = 20 banks: 20*i mod 64 distinct for i = 0..15.
#define VT_STRIDE 40
#define XL_STRIDE 40
#define QL_STRIDE 1032  // 2064 B -> 4-bank shift per row, conflict-free frags

static __device__ __forceinline__ v16h cat16(v8h lo, v8h hi) {
  v16h r;
#pragma unroll
  for (int i = 0; i < 8; ++i) { r[i] = lo[i]; r[i + 8] = hi[i]; }
  return r;
}
static __device__ __forceinline__ v8f zero8() {
  v8f z;
#pragma unroll
  for (int i = 0; i < 8; ++i) z[i] = 0.f;
  return z;
}
static __device__ __forceinline__ v8f wmma16(v16h a, v16h b, v8f c) {
  return __builtin_amdgcn_wmma_f32_16x16x32_f16(false, a, false, b, (short)0, c,
                                                false, false);
}

// ---------------------------------------------------------------------------
// Kernel 1: QKV projection. grid = (M/128, N/128, 3). 256 threads = 8 waves
// (4 M-slices x 2 N-slices); each wave computes 32x64 = 2x4 WMMA tiles.
// ---------------------------------------------------------------------------
__global__ __launch_bounds__(256) void qkv_proj_kernel(
    const float* __restrict__ x, const float* __restrict__ Wq,
    const float* __restrict__ Wk, const float* __restrict__ Wv,
    _Float16* __restrict__ Qh, _Float16* __restrict__ Kh,
    _Float16* __restrict__ Vh) {
  __shared__ _Float16 Xl[128 * XL_STRIDE];  // x tile 128x32 f16 (padded)
  __shared__ _Float16 WT[128 * XL_STRIDE];  // W tile transposed [n][k] (padded)

  const int t = threadIdx.x;
  const int lane = t & 31;
  const int wave = t >> 5;
  const int wm = wave & 3;   // 0..3 -> 32-row slice
  const int wn = wave >> 2;  // 0..1 -> 64-col slice
  const int l15 = lane & 15;
  const int lhi = lane >> 4;
  const int m0 = blockIdx.x * 128;
  const int n0 = blockIdx.y * 128;
  const float* W = (blockIdx.z == 0) ? Wq : (blockIdx.z == 1) ? Wk : Wv;
  _Float16* Out = (blockIdx.z == 0) ? Qh : (blockIdx.z == 1) ? Kh : Vh;

  v8f acc[2][4];
#pragma unroll
  for (int i = 0; i < 2; ++i)
#pragma unroll
    for (int j = 0; j < 4; ++j) acc[i][j] = zero8();

  for (int kt = 0; kt < D_; kt += 32) {
    // Stage x tile (128x32 f32 -> f16), coalesced v4f reads, packed b64 store.
#pragma unroll
    for (int p = 0; p < 4; ++p) {
      int r = p * 32 + (t >> 3);
      int c = (t & 7) * 4;
      v4f v = *(const v4f*)(x + (size_t)(m0 + r) * D_ + kt + c);
      v4h h;
      h[0] = (_Float16)v.x; h[1] = (_Float16)v.y;
      h[2] = (_Float16)v.z; h[3] = (_Float16)v.w;
      *(v4h*)&Xl[r * XL_STRIDE + c] = h;
    }
    // Stage W tile transposed (32x128 f32 -> WT[128][32] f16).
#pragma unroll
    for (int p = 0; p < 4; ++p) {
      int k = p * 8 + (t >> 5);
      int n = (t & 31) * 4;
      v4f v = *(const v4f*)(W + (size_t)(kt + k) * D_ + n0 + n);
      WT[(n + 0) * XL_STRIDE + k] = (_Float16)v.x;
      WT[(n + 1) * XL_STRIDE + k] = (_Float16)v.y;
      WT[(n + 2) * XL_STRIDE + k] = (_Float16)v.z;
      WT[(n + 3) * XL_STRIDE + k] = (_Float16)v.w;
    }
    __syncthreads();

    v16h a[2], b[4];
#pragma unroll
    for (int i = 0; i < 2; ++i) {
      int row = wm * 32 + i * 16 + l15;
      int kb = lhi * 8;
      v8h lo = *(const v8h*)&Xl[row * XL_STRIDE + kb];
      v8h hi = *(const v8h*)&Xl[row * XL_STRIDE + kb + 16];
      a[i] = cat16(lo, hi);
    }
#pragma unroll
    for (int j = 0; j < 4; ++j) {
      int col = wn * 64 + j * 16 + l15;
      int ko = lhi * 8;
      v8h lo = *(const v8h*)&WT[col * XL_STRIDE + ko];
      v8h hi = *(const v8h*)&WT[col * XL_STRIDE + ko + 16];
      b[j] = cat16(lo, hi);
    }
#pragma unroll
    for (int i = 0; i < 2; ++i)
#pragma unroll
      for (int j = 0; j < 4; ++j) acc[i][j] = wmma16(a[i], b[j], acc[i][j]);
    __syncthreads();
  }

#pragma unroll
  for (int i = 0; i < 2; ++i)
#pragma unroll
    for (int j = 0; j < 4; ++j)
#pragma unroll
      for (int r = 0; r < 8; ++r) {
        int row = m0 + wm * 32 + i * 16 + lhi * 8 + r;
        int col = n0 + wn * 64 + j * 16 + l15;
        Out[(size_t)row * D_ + col] = (_Float16)acc[i][j][r];
      }
}

// ---------------------------------------------------------------------------
// Kernel 2: flash-style causal attention. grid = (S/16, B), 256 threads.
// ---------------------------------------------------------------------------
#define SM_QL 0
#define SM_VT 33024    // 16*1032*2
#define SM_SP 114944   // + 1024*40*2
#define SM_SC 131328   // + 8*16*32*4
#define SM_PL 133376   // + 16*32*4
#define SM_M  134400   // + 16*32*2
#define SM_L  134464
#define SM_F  134528
#define SMEM_BYTES 134592

__global__ __launch_bounds__(256) void attn_kernel(
    const _Float16* __restrict__ Qh, const _Float16* __restrict__ Kh,
    const _Float16* __restrict__ Vh, float* __restrict__ Out) {
  extern __shared__ char smem[];
  _Float16* Ql = (_Float16*)(smem + SM_QL);  // [16][QL_STRIDE] f16
  _Float16* Vt = (_Float16*)(smem + SM_VT);  // [1024][VT_STRIDE] f16
  float* Sp = (float*)(smem + SM_SP);        // 8 waves x 16 x 32 f32
  float* Sc = (float*)(smem + SM_SC);        // 16 x 32 f32
  _Float16* Pl = (_Float16*)(smem + SM_PL);  // 16 x 32 f16
  float* mrow = (float*)(smem + SM_M);
  float* lrow = (float*)(smem + SM_L);
  float* frow = (float*)(smem + SM_F);

  const int t = threadIdx.x;
  const int lane = t & 31;
  const int wave = t >> 5;  // 0..7 -> 128-wide d-slice
  const int l15 = lane & 15;
  const int lhi = lane >> 4;
  const int q0 = blockIdx.x * 16;
  const int b = blockIdx.y;
  const size_t baseQ = ((size_t)b * S_ + q0) * D_;
  const size_t baseKV = (size_t)b * S_ * D_;
  const int dbase = wave * 128;

  // Async copy Q tile (16x1024 f16) global -> LDS, no VGPR round trip.
  // GVS form: lds-byte-offset VGPR, 32-bit global offset VGPR, SGPR64 base.
  {
    const _Float16* gsrc = Qh + baseQ;
#pragma unroll
    for (int p = 0; p < 8; ++p) {
      int idx = p * 2048 + t * 8;  // 8 halves per chunk, within one Q row
      int row = idx >> 10;
      int col = idx & 1023;
      unsigned ldsoff = (unsigned)((row * QL_STRIDE + col) * 2);
      unsigned goff = (unsigned)(idx * 2);
      asm volatile("global_load_async_to_lds_b128 %0, %1, %2"
                   :
                   : "v"(ldsoff), "v"(goff), "s"(gsrc)
                   : "memory");
    }
    asm volatile("s_wait_asynccnt 0x0" ::: "memory");
  }
  if (t < 16) { mrow[t] = NEGBIG; lrow[t] = 0.f; }

  v8f acc[8];
#pragma unroll
  for (int j = 0; j < 8; ++j) acc[j] = zero8();

  const int nkb = ((q0 + 15) >> 5) + 1;  // causal: keys 0..q0+15
  __syncthreads();

  for (int kb = 0; kb < nkb; ++kb) {
    const int k0 = kb * 32;

    // Stage V block transposed: Vt[dcol][key], 4-key in-register transpose,
    // packed ds_store_b64 (2-way bank pattern with padded stride).
#pragma unroll
    for (int p = 0; p < 4; ++p) {
      int tau = t + 256 * p;
      int dc0 = (tau & 127) * 8;
      int ka = (tau >> 7) * 4;
      const _Float16* vp = Vh + baseKV + (size_t)(k0 + ka) * D_ + dc0;
      v8h va0 = *(const v8h*)(vp);
      v8h va1 = *(const v8h*)(vp + D_);
      v8h va2 = *(const v8h*)(vp + 2 * D_);
      v8h va3 = *(const v8h*)(vp + 3 * D_);
#pragma unroll
      for (int u = 0; u < 8; ++u) {
        v4h w;
        w[0] = va0[u]; w[1] = va1[u]; w[2] = va2[u]; w[3] = va3[u];
        *(v4h*)&Vt[(dc0 + u) * VT_STRIDE + ka] = w;
      }
    }

    // Partial scores over this wave's 128-wide d slice.
    v8f sacc[2];
    sacc[0] = zero8(); sacc[1] = zero8();
#pragma unroll
    for (int kk = 0; kk < 4; ++kk) {
      int d0 = dbase + kk * 32;
      v8h alo = *(const v8h*)&Ql[l15 * QL_STRIDE + d0 + lhi * 8];
      v8h ahi = *(const v8h*)&Ql[l15 * QL_STRIDE + d0 + lhi * 8 + 16];
      v16h a = cat16(alo, ahi);
#pragma unroll
      for (int j = 0; j < 2; ++j) {
        int key = k0 + j * 16 + l15;
        const _Float16* kp = Kh + baseKV + (size_t)key * D_ + d0 + lhi * 8;
        v8h blo = *(const v8h*)kp;
        v8h bhi = *(const v8h*)(kp + 16);
        sacc[j] = wmma16(a, cat16(blo, bhi), sacc[j]);
      }
    }
#pragma unroll
    for (int j = 0; j < 2; ++j)
#pragma unroll
      for (int r = 0; r < 8; ++r)
        Sp[wave * 512 + (lhi * 8 + r) * 32 + j * 16 + l15] = sacc[j][r];
    __syncthreads();

    // Reduce 8 partials, fold in 1/sqrt(d_k) = 1/32.
#pragma unroll
    for (int e = 0; e < 2; ++e) {
      int idx = e * 256 + t;
      float s = 0.f;
#pragma unroll
      for (int w2 = 0; w2 < 8; ++w2) s += Sp[w2 * 512 + idx];
      Sc[idx] = s * 0.03125f;
    }
    __syncthreads();

    // Online softmax, one thread per query row (reference mask semantics:
    // masked -> -1e9-scale -> exp == 0).
    if (t < 16) {
      int q = q0 + t;
      float sb[32];
      float mx = mrow[t];
#pragma unroll
      for (int c = 0; c < 32; ++c) {
        float s = (k0 + c <= q) ? Sc[t * 32 + c] : NEGBIG;
        sb[c] = s;
        mx = fmaxf(mx, s);
      }
      float f = __expf(mrow[t] - mx);
      float sum = 0.f;
#pragma unroll
      for (int c = 0; c < 32; ++c) {
        float p = (sb[c] <= NEGBIG * 0.5f) ? 0.f : __expf(sb[c] - mx);
        Pl[t * 32 + c] = (_Float16)p;
        sum += p;
      }
      lrow[t] = lrow[t] * f + sum;
      mrow[t] = mx;
      frow[t] = f;
    }
    __syncthreads();

    // O = O*f + P @ V over this wave's d-slice.
    {
      v4f f0 = *(const v4f*)&frow[lhi * 8];
      v4f f1 = *(const v4f*)&frow[lhi * 8 + 4];
      v8h plo = *(const v8h*)&Pl[l15 * 32 + lhi * 8];
      v8h phi = *(const v8h*)&Pl[l15 * 32 + lhi * 8 + 16];
      v16h pa = cat16(plo, phi);
#pragma unroll
      for (int j = 0; j < 8; ++j) {
#pragma unroll
        for (int r = 0; r < 8; ++r) {
          float f = (r < 4) ? f0[r] : f1[r - 4];
          acc[j][r] *= f;
        }
        int dc = dbase + j * 16 + l15;
        v8h blo = *(const v8h*)&Vt[dc * VT_STRIDE + lhi * 8];
        v8h bhi = *(const v8h*)&Vt[dc * VT_STRIDE + lhi * 8 + 16];
        acc[j] = wmma16(pa, cat16(blo, bhi), acc[j]);
      }
    }
    __syncthreads();
  }

  // Normalize by l and store f32 output.
  v4f l0 = *(const v4f*)&lrow[lhi * 8];
  v4f l1 = *(const v4f*)&lrow[lhi * 8 + 4];
#pragma unroll
  for (int j = 0; j < 8; ++j) {
#pragma unroll
    for (int r = 0; r < 8; ++r) {
      float li = (r < 4) ? l0[r] : l1[r - 4];
      int row = q0 + lhi * 8 + r;
      int col = dbase + j * 16 + l15;
      Out[((size_t)b * S_ + row) * D_ + col] = acc[j][r] / li;
    }
  }
}

extern "C" void kernel_launch(void* const* d_in, const int* in_sizes, int n_in,
                              void* d_out, int out_size, void* d_ws,
                              size_t ws_size, hipStream_t stream) {
  (void)in_sizes; (void)n_in; (void)out_size; (void)ws_size;
  const float* x = (const float*)d_in[0];
  const float* Wq = (const float*)d_in[1];
  const float* Wk = (const float*)d_in[2];
  const float* Wv = (const float*)d_in[3];
  float* out = (float*)d_out;

  _Float16* Qh = (_Float16*)d_ws;
  _Float16* Kh = Qh + (size_t)B_ * S_ * D_;
  _Float16* Vh = Kh + (size_t)B_ * S_ * D_;

  (void)hipFuncSetAttribute((const void*)attn_kernel,
                            hipFuncAttributeMaxDynamicSharedMemorySize,
                            SMEM_BYTES);

  dim3 g1((B_ * S_) / 128, D_ / 128, 3);
  qkv_proj_kernel<<<g1, dim3(256), 0, stream>>>(x, Wq, Wk, Wv, Qh, Kh, Vh);

  dim3 g2(S_ / 16, B_);
  attn_kernel<<<g2, dim3(256), SMEM_BYTES, stream>>>(Qh, Kh, Vh, out);
}
